// GRU_52553219834502
// MI455X (gfx1250) — matrix operand order repaired and checked by reading, not compiled, not run on hardware
//
#include <hip/hip_runtime.h>
#include <hip/hip_bf16.h>

// ---------------------------------------------------------------------------
// Fused 2-layer GRU + FC for MI455X (gfx1250), wave32 + WMMA f16->f32,
// async global->LDS staging of the x stream (double-buffered per timestep).
// B=65536, T=5, D=52, H=40, C=22. Each wave owns 16 batch rows.
// ---------------------------------------------------------------------------

typedef __attribute__((ext_vector_type(16))) _Float16 v16h;
typedef __attribute__((ext_vector_type(8)))  _Float16 v8h;
typedef __attribute__((ext_vector_type(8)))  float    v8f;

// exact pointer types for the async-LDS builtin (from hipcc's diagnostic:
// param is 'int __vector(4) __device__ *' i.e. AS1 int4*, LDS side AS3 int4*)
typedef int v4i_gnu __attribute__((vector_size(16)));
typedef __attribute__((address_space(1))) v4i_gnu* as1_v4i_ptr;
typedef __attribute__((address_space(3))) v4i_gnu* as3_v4i_ptr;

#define GB   65536
#define GT   5
#define GD   52
#define GH   40
#define GC   22
#define G3H  120

// padded dims
#define KP_X   64    // D=52 -> 64   (2 k-tiles)
#define KP_H   64    // H=40 -> 64   (2 k-tiles)
#define NP_G   128   // 3H=120 -> 128 (8 n-tiles)
#define KP_FC  224   // T*H=200 -> 224 (7 k-tiles)
#define NP_FC  32    // C=22 -> 32   (2 n-tiles)

// f16 weight image offsets inside d_ws (in halves)
#define WOFF_WIH0  0
#define WOFF_WHH0  8192      // 128*64
#define WOFF_WIH1  16384
#define WOFF_WHH1  24576
#define WOFF_WFC   32768     // 32*224 = 7168
#define WTOTAL     39936

// per-wave LDS layout (bytes)
#define L_XT    0        // x_t staged   16x64 f16   = 2048
#define L_HCUR  2048     // h current    16x64 f16   = 2048
#define L_H1    4096     // layer1 in    16x5x64 f16 = 10240
#define L_H2F   14336    // flat h2      16x224 f16  = 7168
#define L_XG    21504    // xg pre-act   16x128 f32  = 8192
#define L_HG    29696    // hg pre-act   16x128 f32  = 8192
#define L_RAW0  37888    // raw x dbl buf 16x56 f32  = 3584
#define L_RAW1  41472    //               16x56 f32  = 3584
#define L_WAVE  45056    // total per wave (16B aligned)
#define RAW_STRIDE 56    // floats per raw row (52 padded to 14 x float4)
#define WAVES_PER_BLOCK 4

// ---- async global->LDS support (gfx1250) ----------------------------------
#if defined(__has_builtin)
#if __has_builtin(__builtin_amdgcn_global_load_async_to_lds_b128)
#define HAVE_ASYNC_LDS 1
#endif
#endif
#ifndef HAVE_ASYNC_LDS
#define HAVE_ASYNC_LDS 0
#endif

#if HAVE_ASYNC_LDS
#if __has_builtin(__builtin_amdgcn_s_wait_asynccnt)
#define WAIT_ASYNC0() __builtin_amdgcn_s_wait_asynccnt(0)
#else
#define WAIT_ASYNC0() asm volatile("s_wait_asynccnt 0" ::: "memory")
#endif
#else
#define WAIT_ASYNC0()
#endif

__device__ __forceinline__ v8f wmma16(v16h a, v16h b, v8f c) {
  return __builtin_amdgcn_wmma_f32_16x16x32_f16(false, a, false, b,
                                                (short)0, c, false, false);
}

__device__ __forceinline__ v16h cat16(v8h lo, v8h hi) {
  return __builtin_shufflevector(lo, hi, 0, 1, 2, 3, 4, 5, 6, 7,
                                 8, 9, 10, 11, 12, 13, 14, 15);
}

// A-fragment (16x32 f16, M x K) from an LDS tile, row-major [16][strideH] halves.
// ISA layout: lane m=L%16 g=L/16; elems 0..7 -> K=32p+8g+e ; 8..15 -> +16
__device__ __forceinline__ v16h load_a(const _Float16* base, int strideH,
                                       int p, int lane) {
  int m = lane & 15, g = lane >> 4;
  int K0 = 32 * p + 8 * g;
  v8h lo = *(const v8h*)(base + m * strideH + K0);
  v8h hi = *(const v8h*)(base + m * strideH + K0 + 16);
  return cat16(lo, hi);
}

// B-fragment (32x16 f16, K x N) from global padded row-major weights W[n][Kpad].
// ISA layout: lane n=L%16 (cols), g=L/16; elems 0..7 -> K=32p+8g+e ; 8..15 -> +16
__device__ __forceinline__ v16h load_b(const _Float16* W, int Kpad,
                                       int j, int p, int lane) {
  int n = 16 * j + (lane & 15);
  int Kb = 32 * p + 8 * (lane >> 4);
  const v8h* q = (const v8h*)(W + n * Kpad + Kb);
  return cat16(q[0], q[2]);
}

__device__ __forceinline__ float sigf(float x) {
  return 1.0f / (1.0f + __expf(-x));
}
__device__ __forceinline__ float tanh_fast(float x) {
  float e = __expf(-2.0f * x);
  return (1.0f - e) / (1.0f + e);
}

// Stage 16 rows x 52 floats of x[:, t, :] into a raw LDS buffer.
// 13 x b128 chunks per row, all 16B aligned (row pitch 208 B, LDS pitch 224 B).
__device__ __forceinline__ void stage_x_async(const float* __restrict__ x,
                                              int b0, int t, float* raw,
                                              int lane) {
  for (int c = lane; c < 16 * 13; c += 32) {
    int m = c / 13, q = c - m * 13;
    const float* g = x + ((size_t)(b0 + m) * GT + t) * GD + q * 4;
    float* l = raw + m * RAW_STRIDE + q * 4;
#if HAVE_ASYNC_LDS
    __builtin_amdgcn_global_load_async_to_lds_b128(
        (as1_v4i_ptr)g, (as3_v4i_ptr)l, 0, 0);
#else
    *(float4*)l = *(const float4*)g;
#endif
  }
}

// ---------------------------------------------------------------------------
// Kernel 0: convert f32 weights to zero-padded f16 images in d_ws.
// ---------------------------------------------------------------------------
__global__ void gru_prep_weights(const float* __restrict__ Wih0,
                                 const float* __restrict__ Whh0,
                                 const float* __restrict__ Wih1,
                                 const float* __restrict__ Whh1,
                                 const float* __restrict__ Wfc,
                                 _Float16* __restrict__ wp) {
  int i = blockIdx.x * blockDim.x + threadIdx.x;
  if (i >= WTOTAL) return;
  float v = 0.0f;
  if (i < WOFF_WHH0) {
    int r = i - WOFF_WIH0, n = r >> 6, k = r & 63;
    if (n < G3H && k < GD) v = Wih0[n * GD + k];
  } else if (i < WOFF_WIH1) {
    int r = i - WOFF_WHH0, n = r >> 6, k = r & 63;
    if (n < G3H && k < GH) v = Whh0[n * GH + k];
  } else if (i < WOFF_WHH1) {
    int r = i - WOFF_WIH1, n = r >> 6, k = r & 63;
    if (n < G3H && k < GH) v = Wih1[n * GH + k];
  } else if (i < WOFF_WFC) {
    int r = i - WOFF_WHH1, n = r >> 6, k = r & 63;
    if (n < G3H && k < GH) v = Whh1[n * GH + k];
  } else {
    int r = i - WOFF_WFC, n = r / KP_FC, k = r % KP_FC;
    if (n < GC && k < GT * GH) v = Wfc[n * (GT * GH) + k];
  }
  wp[i] = (_Float16)v;
}

// ---------------------------------------------------------------------------
// Kernel 1: fused GRU(2 layers) + FC. One wave = 16 batch rows.
// ---------------------------------------------------------------------------
__global__ void __launch_bounds__(128)
gru_fused(const float* __restrict__ x,
          const float* __restrict__ bih0, const float* __restrict__ bhh0,
          const float* __restrict__ bih1, const float* __restrict__ bhh1,
          const float* __restrict__ bfc,
          const _Float16* __restrict__ wp,
          float* __restrict__ out) {
  extern __shared__ char smem[];
  const int lane = threadIdx.x & 31;
  const int wave = threadIdx.x >> 5;
  const int b0 = (blockIdx.x * WAVES_PER_BLOCK + wave) * 16;

  char* wb = smem + (size_t)wave * L_WAVE;
  _Float16* x_t  = (_Float16*)(wb + L_XT);
  _Float16* hcur = (_Float16*)(wb + L_HCUR);
  _Float16* h1t  = (_Float16*)(wb + L_H1);
  _Float16* h2f  = (_Float16*)(wb + L_H2F);
  float*    xg   = (float*)(wb + L_XG);
  float*    hg   = (float*)(wb + L_HG);
  float*    raw0 = (float*)(wb + L_RAW0);
  float*    raw1 = (float*)(wb + L_RAW1);

  // zero hcur + h1t + h2f (covers all K/N padding lanes), contiguous range
  {
    float4 z4 = make_float4(0.f, 0.f, 0.f, 0.f);
    float4* zp = (float4*)(wb + L_HCUR);
    for (int i = lane; i < (L_XG - L_HCUR) / 16; i += 32) zp[i] = z4;
  }

  const _Float16* wih0 = wp + WOFF_WIH0;
  const _Float16* whh0 = wp + WOFF_WHH0;
  const _Float16* wih1 = wp + WOFF_WIH1;
  const _Float16* whh1 = wp + WOFF_WHH1;
  const _Float16* wfc  = wp + WOFF_WFC;

  // kick off async staging of x[:, 0, :]
  stage_x_async(x, b0, 0, raw0, lane);

  for (int layer = 0; layer < 2; ++layer) {
    const _Float16* Wi = layer ? wih1 : wih0;
    const _Float16* Wh = layer ? whh1 : whh0;
    const float* bi = layer ? bih1 : bih0;
    const float* bh = layer ? bhh1 : bhh0;

    // reset h = 0 for this layer (2048 B = 128 float4)
    {
      float4 z4 = make_float4(0.f, 0.f, 0.f, 0.f);
      float4* hz = (float4*)hcur;
      for (int i = lane; i < 128; i += 32) hz[i] = z4;
    }

    for (int t = 0; t < GT; ++t) {
      const _Float16* Ain;
      int strideA;
      if (layer == 0) {
        // wait for async copy of this timestep, convert f32 -> padded f16
        WAIT_ASYNC0();
        const float* rawc = (t & 1) ? raw1 : raw0;
        for (int i = lane; i < 16 * KP_X; i += 32) {
          int m = i >> 6, d = i & 63;
          float v = (d < GD) ? rawc[m * RAW_STRIDE + d] : 0.0f;
          x_t[i] = (_Float16)v;
        }
        // prefetch next timestep into the other raw buffer (behind WMMAs)
        if (t + 1 < GT)
          stage_x_async(x, b0, t + 1, (t & 1) ? raw0 : raw1, lane);
        Ain = x_t; strideA = KP_X;
      } else {
        Ain = h1t + t * KP_H; strideA = GT * KP_H;  // 320 halves/row
      }

      // accumulators seeded with biases (per-column constant per lane)
      v8f accx[8], acch[8];
#pragma unroll
      for (int j = 0; j < 8; ++j) {
        int c = 16 * j + (lane & 15);
        float bx = (c < G3H) ? bi[c] : 0.0f;
        float bhv = (c < G3H) ? bh[c] : 0.0f;
#pragma unroll
        for (int r = 0; r < 8; ++r) { accx[j][r] = bx; acch[j][r] = bhv; }
      }

      v16h a0 = load_a(Ain, strideA, 0, lane);
      v16h a1 = load_a(Ain, strideA, 1, lane);
      v16h g0 = load_a(hcur, KP_H, 0, lane);
      v16h g1 = load_a(hcur, KP_H, 1, lane);

#pragma unroll
      for (int j = 0; j < 8; ++j) {
        accx[j] = wmma16(a0, load_b(Wi, KP_X, j, 0, lane), accx[j]);
        accx[j] = wmma16(a1, load_b(Wi, KP_X, j, 1, lane), accx[j]);
        acch[j] = wmma16(g0, load_b(Wh, KP_H, j, 0, lane), acch[j]);
        acch[j] = wmma16(g1, load_b(Wh, KP_H, j, 1, lane), acch[j]);
      }

      // spill D-layout accumulators to LDS (lane holds col c, rows r+8*(L/16))
#pragma unroll
      for (int j = 0; j < 8; ++j) {
        int c = 16 * j + (lane & 15);
#pragma unroll
        for (int r = 0; r < 8; ++r) {
          int m = r + 8 * (lane >> 4);
          xg[m * NP_G + c] = accx[j][r];
          hg[m * NP_G + c] = acch[j][r];
        }
      }

      // elementwise GRU update (PyTorch order r,z,n); 640 cells / 32 lanes
      for (int idx = lane; idx < 16 * GH; idx += 32) {
        int m = idx / GH, n = idx - m * GH;
        float rg = sigf(xg[m * NP_G + n]      + hg[m * NP_G + n]);
        float zg = sigf(xg[m * NP_G + GH + n] + hg[m * NP_G + GH + n]);
        float ng = tanh_fast(xg[m * NP_G + 2 * GH + n] +
                             rg * hg[m * NP_G + 2 * GH + n]);
        float hp = (float)hcur[m * KP_H + n];
        float hn = (1.0f - zg) * ng + zg * hp;
        hcur[m * KP_H + n] = (_Float16)hn;
        if (layer == 0) h1t[m * (GT * KP_H) + t * KP_H + n] = (_Float16)hn;
        else            h2f[m * KP_FC + t * GH + n]         = (_Float16)hn;
      }
    }
  }

  // FC: out[16,22] = h2f[16,200] @ Wfc^T + bfc   (7 k-tiles, 2 n-tiles)
  v8f acc[2];
#pragma unroll
  for (int j = 0; j < 2; ++j) {
    int c = 16 * j + (lane & 15);
    float bv = (c < GC) ? bfc[c] : 0.0f;
#pragma unroll
    for (int r = 0; r < 8; ++r) acc[j][r] = bv;
  }
#pragma unroll
  for (int p = 0; p < 7; ++p) {
    v16h a = load_a(h2f, KP_FC, p, lane);
    acc[0] = wmma16(a, load_b(wfc, KP_FC, 0, p, lane), acc[0]);
    acc[1] = wmma16(a, load_b(wfc, KP_FC, 1, p, lane), acc[1]);
  }
#pragma unroll
  for (int j = 0; j < 2; ++j) {
    int c = 16 * j + (lane & 15);
    if (c < GC) {
#pragma unroll
      for (int r = 0; r < 8; ++r) {
        int m = r + 8 * (lane >> 4);
        out[(size_t)(b0 + m) * GC + c] = acc[j][r];
      }
    }
  }
}

// ---------------------------------------------------------------------------
extern "C" void kernel_launch(void* const* d_in, const int* in_sizes, int n_in,
                              void* d_out, int out_size, void* d_ws,
                              size_t ws_size, hipStream_t stream) {
  const float* x    = (const float*)d_in[0];
  const float* Wih0 = (const float*)d_in[1];
  const float* Whh0 = (const float*)d_in[2];
  const float* bih0 = (const float*)d_in[3];
  const float* bhh0 = (const float*)d_in[4];
  const float* Wih1 = (const float*)d_in[5];
  const float* Whh1 = (const float*)d_in[6];
  const float* bih1 = (const float*)d_in[7];
  const float* bhh1 = (const float*)d_in[8];
  const float* Wfc  = (const float*)d_in[9];
  const float* bfc  = (const float*)d_in[10];
  _Float16* wp = (_Float16*)d_ws;
  float* out = (float*)d_out;

  gru_prep_weights<<<(WTOTAL + 255) / 256, 256, 0, stream>>>(
      Wih0, Whh0, Wih1, Whh1, Wfc, wp);

  const int blocks = GB / (16 * WAVES_PER_BLOCK);  // 1024
  const size_t shmem = (size_t)WAVES_PER_BLOCK * L_WAVE;  // ~176 KB dynamic LDS
  gru_fused<<<blocks, 128, shmem, stream>>>(
      x, bih0, bhh0, bih1, bhh1, bfc, wp, out);
}
